// Attention_10453950398722
// MI455X (gfx1250) — compile-verified, matrix operand order checked by reading, not run on hardware
//
#include <hip/hip_runtime.h>
#include <hip/hip_bf16.h>

// ---------------------------------------------------------------------------
// Multi-head attention (B=8, N=1024, C=768, H=12, D=64) for gfx1250.
// All GEMM work on v_wmma_f32_16x16x32_bf16; softmax on VALU.
// ---------------------------------------------------------------------------

#define BATCH 8
#define SEQ   1024
#define CDIM  768
#define NH    12
#define HD    64
#define SCALE 0.125f          // 64^-0.5

typedef __bf16 bf16_t;
typedef __attribute__((ext_vector_type(16))) __bf16 v16bf;
typedef __attribute__((ext_vector_type(8)))  __bf16 v8bf;   // 16-byte chunk
typedef __attribute__((ext_vector_type(8)))  float  v8f;

// A-fragment (and B^T-in-A-layout fragment) loader for 16x16x32 bf16 WMMA.
// Per ISA 7.12.2 (16-bit A 16x32): lane = row (mod 16); lane[4] selects the
// K = {0..7,16..23} vs {8..15,24..31} halves.  Each lane: two b128 loads,
// K-contiguous.  `p0` points at row 0 of the fragment, `ld` = row stride in
// elements, `k0` = K offset (multiple of 32).
template <typename PT>
__device__ __forceinline__ v16bf load_frag(PT p0, int ld, int k0, int lane) {
    int r   = lane & 15;
    int hlf = lane >> 4;
    PT p = p0 + (size_t)r * ld + k0 + hlf * 8;
    v8bf lo = *(const v8bf*)(p);        // K = k0 + hlf*8 + [0..7]
    v8bf hi = *(const v8bf*)(p + 16);   // K = k0 + 16 + hlf*8 + [0..7]
    v16bf out;
#pragma unroll
    for (int i = 0; i < 8; ++i) { out[i] = lo[i]; out[i + 8] = hi[i]; }
    return out;
}

__device__ __forceinline__ v8f wmma_bf16(v16bf a, v16bf b, v8f c) {
    return __builtin_amdgcn_wmma_f32_16x16x32_bf16(
        false, a, false, b, (short)0, c, false, false);
}

// ---------------------------------------------------------------------------
// K0: f32 -> bf16 conversion (grid-stride)
// ---------------------------------------------------------------------------
__global__ void cvt_f32_bf16(const float* __restrict__ src,
                             bf16_t* __restrict__ dst, int n) {
    int i = blockIdx.x * blockDim.x + threadIdx.x;
    int stride = gridDim.x * blockDim.x;
    for (; i < n; i += stride) dst[i] = (bf16_t)src[i];
}

// ---------------------------------------------------------------------------
// K1: QKV GEMM.  out[m, ncol] = x[m,:] . qkv_w[ncol,:] + qkv_b[ncol]
// m in [0,8192), ncol in [0,2304).  Each wave: 32x64 tile (2 A-frags x 4
// B-tiles -> 8 WMMA : 12 b128 loads per k-step).
// Scatters bf16 results into Q[bh][n][d], K[bh][n][d], Vt[bh][d][n].
// grid = (256, 9), block = 128 (4 waves)
// ---------------------------------------------------------------------------
__global__ void qkv_gemm(const bf16_t* __restrict__ xb,
                         const bf16_t* __restrict__ wb,
                         const float*  __restrict__ bias,
                         bf16_t* __restrict__ Qb,
                         bf16_t* __restrict__ Kb,
                         bf16_t* __restrict__ Vt) {
    int lane = threadIdx.x & 31;
    int wave = threadIdx.x >> 5;
    int m0 = blockIdx.x * 32;
    int n0 = (blockIdx.y * 4 + wave) * 64;

    v8f acc[2][4];
#pragma unroll
    for (int i = 0; i < 2; ++i)
#pragma unroll
        for (int t = 0; t < 4; ++t) acc[i][t] = v8f{};

    for (int k = 0; k < CDIM; k += 32) {
        v16bf a0 = load_frag(xb + (size_t)m0 * CDIM,        CDIM, k, lane);
        v16bf a1 = load_frag(xb + (size_t)(m0 + 16) * CDIM, CDIM, k, lane);
#pragma unroll
        for (int t = 0; t < 4; ++t) {
            v16bf b = load_frag(wb + (size_t)(n0 + t * 16) * CDIM, CDIM, k, lane);
            acc[0][t] = wmma_bf16(a0, b, acc[0][t]);
            acc[1][t] = wmma_bf16(a1, b, acc[1][t]);
        }
    }

    int r   = lane & 15;
    int hlf = lane >> 4;
#pragma unroll
    for (int t = 0; t < 4; ++t) {
        int ncol  = n0 + t * 16 + r;          // column (output feature)
        int which = ncol / CDIM;              // 0=q 1=k 2=v (uniform per tile)
        int c     = ncol % CDIM;
        int h     = c >> 6;
        int d     = c & 63;
        float bv  = bias[ncol];
#pragma unroll
        for (int mi = 0; mi < 2; ++mi) {
#pragma unroll
            for (int reg = 0; reg < 8; ++reg) {
                int m  = m0 + mi * 16 + hlf * 8 + reg;  // row (token)
                int bb = m >> 10;                       // batch
                int nn = m & (SEQ - 1);                 // seq pos
                bf16_t v = (bf16_t)(acc[mi][t][reg] + bv);
                size_t bh = (size_t)bb * NH + h;
                if (which == 0)      Qb[(bh * SEQ + nn) * HD + d] = v;
                else if (which == 1) Kb[(bh * SEQ + nn) * HD + d] = v;
                else                 Vt[(bh * HD + d) * SEQ + nn] = v;
            }
        }
    }
}

// ---------------------------------------------------------------------------
// K2: attention.  One block per (bh, 32-query-row strip).
//   Phase A: S[32][1024] = Q Kt * SCALE  (WMMA into LDS f32; Q-frag reuse x4)
//   Phase B: row softmax (8 threads per row), P[32][1024] bf16 in LDS
//   Phase C: O[32][64] = P @ V  (WMMA; A from LDS, B = Vt from global)
// grid = 8*12*32 = 3072, block = 256 (8 waves), dyn-LDS = 197888 B
// ---------------------------------------------------------------------------
#define ATTN_LDS_BYTES (32 * 1024 * 4 + 32 * 1024 * 2 + 32 * 8 * 4 + 64 * 4)

__global__ void attn_kernel(const bf16_t* __restrict__ Qb,
                            const bf16_t* __restrict__ Kb,
                            const bf16_t* __restrict__ Vt,
                            bf16_t* __restrict__ ctx) {
    extern __shared__ char smem[];
    float*  S      = (float*)smem;                           // 32*1024 f32
    bf16_t* P      = (bf16_t*)(smem + 32 * 1024 * 4);        // 32*1024 bf16
    float*  red    = (float*)(smem + 32 * 1024 * 4 + 32 * 1024 * 2); // [32][8]
    float*  rowst  = red + 32 * 8;                           // max[32], sum[32]

    int tid  = threadIdx.x;
    int lane = tid & 31;
    int wave = tid >> 5;

    int bh   = blockIdx.x >> 5;          // b*NH + h
    int rb   = blockIdx.x & 31;
    int row0 = rb * 32;

    const bf16_t* Qhead = Qb + (size_t)bh * SEQ * HD;
    const bf16_t* Khead = Kb + (size_t)bh * SEQ * HD;
    const bf16_t* Vhead = Vt + (size_t)bh * HD * SEQ;

    int r   = lane & 15;
    int hlf = lane >> 4;

    // ---- Phase A: scores ------------------------------------------------
    // 2 row-tiles x 16 strips (4 col-tiles each) = 32 strips over 8 waves.
    // Q fragments loaded once per strip and reused for 4 K-tiles.
    for (int strip = wave; strip < 32; strip += 8) {
        int mt  = strip & 1;
        int nt0 = (strip >> 1) * 4;
        const bf16_t* qrow = Qhead + (size_t)(row0 + mt * 16) * HD;
        v16bf a0 = load_frag(qrow, HD, 0,  lane);
        v16bf a1 = load_frag(qrow, HD, 32, lane);
#pragma unroll
        for (int t = 0; t < 4; ++t) {
            int nt = nt0 + t;
            const bf16_t* krow = Khead + (size_t)(nt * 16) * HD;
            v8f acc = v8f{};
            acc = wmma_bf16(a0, load_frag(krow, HD, 0,  lane), acc);
            acc = wmma_bf16(a1, load_frag(krow, HD, 32, lane), acc);
#pragma unroll
            for (int reg = 0; reg < 8; ++reg)
                S[(mt * 16 + hlf * 8 + reg) * 1024 + nt * 16 + r] =
                    acc[reg] * SCALE;
        }
    }
    __syncthreads();

    // ---- Phase B: softmax ----------------------------------------------
    {
        int row = tid >> 3;              // 32 rows
        int seg = tid & 7;               // 8 segments x 128 cols
        float* Srow = S + row * 1024 + seg * 128;

        float m = -1e30f;
        for (int i = 0; i < 128; ++i) m = fmaxf(m, Srow[i]);
        red[row * 8 + seg] = m;
        __syncthreads();
        if (seg == 0) {
            float mm = red[row * 8];
            for (int j = 1; j < 8; ++j) mm = fmaxf(mm, red[row * 8 + j]);
            rowst[row] = mm;             // row max
        }
        __syncthreads();
        float mm = rowst[row];
        float s = 0.f;
        for (int i = 0; i < 128; ++i) {
            float e = __expf(Srow[i] - mm);
            Srow[i] = e;
            s += e;
        }
        __syncthreads();
        red[row * 8 + seg] = s;
        __syncthreads();
        if (seg == 0) {
            float ss = 0.f;
            for (int j = 0; j < 8; ++j) ss += red[row * 8 + j];
            rowst[32 + row] = ss;        // row sum
        }
        __syncthreads();
        float inv = 1.0f / rowst[32 + row];
        bf16_t* Prow = P + row * 1024 + seg * 128;
        for (int i = 0; i < 128; ++i) Prow[i] = (bf16_t)(Srow[i] * inv);
    }
    __syncthreads();

    // ---- Phase C: O = P @ V --------------------------------------------
    // 2 row-tiles x 4 d-tiles = 8 tiles -> one per wave; K loop over 1024.
    {
        int mt = wave >> 2;              // 0..1
        int dt = wave & 3;               // 0..3
        v8f acc = v8f{};
        for (int kk = 0; kk < SEQ; kk += 32) {
            v16bf a = load_frag(P + (size_t)(mt * 16) * 1024, 1024, kk, lane); // LDS
            v16bf b = load_frag(Vhead + (size_t)(dt * 16) * SEQ, SEQ, kk, lane);
            acc = wmma_bf16(a, b, acc);
        }
        int h = bh % NH, b = bh / NH;
#pragma unroll
        for (int reg = 0; reg < 8; ++reg) {
            int n = row0 + mt * 16 + hlf * 8 + reg;
            ctx[((size_t)b * SEQ + n) * CDIM + h * 64 + dt * 16 + r] =
                (bf16_t)acc[reg];
        }
    }
}

// ---------------------------------------------------------------------------
// K3: output projection.  out[m, ncol] = ctx[m,:] . proj_w[ncol,:] + proj_b
// M=8192, K=768, N=768; wave computes 32x64.  grid=(256,3), block=128.
// ---------------------------------------------------------------------------
__global__ void proj_gemm(const bf16_t* __restrict__ ctx,
                          const bf16_t* __restrict__ wb,
                          const float*  __restrict__ bias,
                          float* __restrict__ out) {
    int lane = threadIdx.x & 31;
    int wave = threadIdx.x >> 5;
    int m0 = blockIdx.x * 32;
    int n0 = (blockIdx.y * 4 + wave) * 64;

    v8f acc[2][4];
#pragma unroll
    for (int i = 0; i < 2; ++i)
#pragma unroll
        for (int t = 0; t < 4; ++t) acc[i][t] = v8f{};

    for (int k = 0; k < CDIM; k += 32) {
        v16bf a0 = load_frag(ctx + (size_t)m0 * CDIM,        CDIM, k, lane);
        v16bf a1 = load_frag(ctx + (size_t)(m0 + 16) * CDIM, CDIM, k, lane);
#pragma unroll
        for (int t = 0; t < 4; ++t) {
            v16bf b = load_frag(wb + (size_t)(n0 + t * 16) * CDIM, CDIM, k, lane);
            acc[0][t] = wmma_bf16(a0, b, acc[0][t]);
            acc[1][t] = wmma_bf16(a1, b, acc[1][t]);
        }
    }

    int r = lane & 15, hlf = lane >> 4;
#pragma unroll
    for (int t = 0; t < 4; ++t) {
        int ncol = n0 + t * 16 + r;
        float bv = bias[ncol];
#pragma unroll
        for (int mi = 0; mi < 2; ++mi) {
#pragma unroll
            for (int reg = 0; reg < 8; ++reg) {
                int m = m0 + mi * 16 + hlf * 8 + reg;
                out[(size_t)m * CDIM + ncol] = acc[mi][t][reg] + bv;
            }
        }
    }
}

// ---------------------------------------------------------------------------
extern "C" void kernel_launch(void* const* d_in, const int* in_sizes, int n_in,
                              void* d_out, int out_size, void* d_ws, size_t ws_size,
                              hipStream_t stream) {
    const float* x      = (const float*)d_in[0];   // [8,1024,768]
    const float* qkv_w  = (const float*)d_in[1];   // [2304,768]
    const float* qkv_b  = (const float*)d_in[2];   // [2304]
    const float* proj_w = (const float*)d_in[3];   // [768,768]
    const float* proj_b = (const float*)d_in[4];   // [768]
    float* out = (float*)d_out;

    char* ws = (char*)d_ws;
    size_t off = 0;
    auto alloc = [&](size_t bytes) -> void* {
        off = (off + 255) & ~(size_t)255;
        void* p = ws + off;
        off += bytes;
        return p;
    };

    const size_t M = (size_t)BATCH * SEQ;                 // 8192
    bf16_t* xb    = (bf16_t*)alloc(M * CDIM * 2);         // x bf16
    bf16_t* qw_b  = (bf16_t*)alloc((size_t)3 * CDIM * CDIM * 2);
    bf16_t* pw_b  = (bf16_t*)alloc((size_t)CDIM * CDIM * 2);
    bf16_t* Qb    = (bf16_t*)alloc((size_t)BATCH * NH * SEQ * HD * 2);
    bf16_t* Kb    = (bf16_t*)alloc((size_t)BATCH * NH * SEQ * HD * 2);
    bf16_t* Vt    = (bf16_t*)alloc((size_t)BATCH * NH * HD * SEQ * 2);
    bf16_t* ctx   = (bf16_t*)alloc(M * CDIM * 2);
    (void)ws_size; (void)n_in; (void)in_sizes; (void)out_size;

    // K0: conversions
    cvt_f32_bf16<<<2048, 256, 0, stream>>>(x,      xb,   (int)(M * CDIM));
    cvt_f32_bf16<<<1024, 256, 0, stream>>>(qkv_w,  qw_b, 3 * CDIM * CDIM);
    cvt_f32_bf16<<<512,  256, 0, stream>>>(proj_w, pw_b, CDIM * CDIM);

    // K1: QKV projection -> Q, K, Vt (bf16)
    {
        dim3 grid(M / 32, (3 * CDIM) / (4 * 64));         // (256, 9)
        qkv_gemm<<<grid, 128, 0, stream>>>(xb, qw_b, qkv_b, Qb, Kb, Vt);
    }

    // K2: attention -> ctx (bf16)
    {
        dim3 grid(BATCH * NH * (SEQ / 32));               // 3072
        attn_kernel<<<grid, 256, ATTN_LDS_BYTES, stream>>>(Qb, Kb, Vt, ctx);
    }

    // K3: output projection -> out (f32)
    {
        dim3 grid(M / 32, CDIM / (4 * 64));               // (256, 3)
        proj_gemm<<<grid, 128, 0, stream>>>(ctx, pw_b, proj_b, out);
    }
}
